// NonTimestepResidualAttentionNorm_11922829214327
// MI455X (gfx1250) — matrix operand order repaired and checked by reading, not compiled, not run on hardware
//
#include <hip/hip_runtime.h>
#include <hip/hip_bf16.h>

typedef __attribute__((ext_vector_type(16))) __bf16 bf16x16;
typedef __attribute__((ext_vector_type(8)))  __bf16 bf16x8;
typedef __attribute__((ext_vector_type(4)))  __bf16 bf16x4;
typedef __attribute__((ext_vector_type(8)))  float  f32x8;
typedef __attribute__((ext_vector_type(4)))  unsigned int u32x4;
typedef __attribute__((ext_vector_type(8)))  int i32x8;
typedef __attribute__((ext_vector_type(4)))  int i32x4;

#define C_DIM 1024
#define T_DIM 1024
#define H_DIM 16

__device__ __forceinline__ f32x8 wmma_bf16(bf16x16 a, bf16x16 b, f32x8 c) {
  return __builtin_amdgcn_wmma_f32_16x16x32_bf16(false, a, false, b, (short)0, c,
                                                 false, false);
}

// Issue a 2-D TENSOR_LOAD_TO_LDS: tile (tileK x tileM), row stride strideK
// elements of 2 bytes, global tile start gaddr -> contiguous LDS at lds_byte.
__device__ __forceinline__ void tdm_load_2d_bf16(unsigned lds_byte,
                                                 const __bf16* gsrc, int dimK,
                                                 int dimM, int tileK, int tileM,
                                                 int strideK) {
  unsigned long long ga = (unsigned long long)(uintptr_t)gsrc;
  u32x4 g0;
  g0[0] = 1u;                                    // count=1, user descriptor
  g0[1] = lds_byte;                              // lds_addr
  g0[2] = (unsigned)(ga & 0xffffffffu);          // global_addr[31:0]
  g0[3] = (unsigned)((ga >> 32) & 0x01ffffffu) | (2u << 30);  // addr[56:32],type=2
  i32x8 g1;
  g1[0] = 0x00010000;                            // data_size=2B, no flags
  g1[1] = (int)((unsigned)(dimK & 0xffff) << 16);           // tensor_dim0[15:0]
  g1[2] = (int)(((unsigned)dimK >> 16) | ((unsigned)(dimM & 0xffff) << 16));
  g1[3] = (int)(((unsigned)dimM >> 16) | ((unsigned)tileK << 16));  // tile_dim0
  g1[4] = tileM;                                 // tile_dim1 (tile_dim2=0)
  g1[5] = strideK;                               // tensor_dim0_stride[31:0]
  g1[6] = 0;
  g1[7] = 0;
  i32x4 z4 = {0, 0, 0, 0};
#if defined(__clang_major__) && __clang_major__ >= 23
  i32x8 z8 = {0, 0, 0, 0, 0, 0, 0, 0};
  __builtin_amdgcn_tensor_load_to_lds(g0, g1, z4, z4, z8, 0);
#else
  __builtin_amdgcn_tensor_load_to_lds(g0, g1, z4, z4, 0);
#endif
}

// ---------------- weight convert: (Cout,Cin,taps) f32 -> (taps,Cout,Cin) bf16
__global__ void wcvt_kernel(const float* __restrict__ w, __bf16* __restrict__ out,
                            int Cout, int Cin, int taps) {
  size_t total = (size_t)Cout * Cin * taps;
  size_t i = (size_t)blockIdx.x * 256 + threadIdx.x;
  if (i >= total) return;
  int tp = (int)(i % taps);
  size_t mc = i / taps;
  out[(size_t)tp * Cout * Cin + mc] = (__bf16)w[i];
}

// ---------------- relative position bias table: tab[h][d], d = (t-s)+1024
__global__ void btab_kernel(const float* __restrict__ emb, float* __restrict__ tab) {
  int i = blockIdx.x * 256 + threadIdx.x;
  if (i >= H_DIM * 2048) return;
  int h = i >> 11, d = i & 2047;
  int delta = d - 1024;
  int n = delta < 0 ? -delta : delta;
  int bucket;
  if (n < 8) {
    bucket = n;
  } else {
    int v = 8 + (int)(__logf((float)n * 0.125f) * (8.0f / __logf(8.0f)));
    bucket = v > 15 ? 15 : v;
  }
  if (delta < 0) bucket += 16;
  tab[i] = emb[bucket * H_DIM + h] * 8.0f;  // REL_SCALE = sqrt(64)
}

// ---------------- GroupNorm (+optional SiLU), fp32 in, fp32 and/or bf16 out
__global__ __launch_bounds__(256) void gn_kernel(
    const float* __restrict__ x, const float* __restrict__ gamma,
    const float* __restrict__ beta, float* __restrict__ outF,
    __bf16* __restrict__ outB, int C, int T, int G, int doSilu) {
  int g = blockIdx.x, b = blockIdx.y;
  int cpg = C / G;
  size_t n = (size_t)cpg * T;
  size_t base = ((size_t)b * C + (size_t)g * cpg) * T;
  const float* p = x + base;
  int tid = threadIdx.x;
  float s = 0.f, ss = 0.f;
  for (size_t i = (size_t)tid * 4; i < n; i += 1024) {
    float4 v = *(const float4*)(p + i);
    s += v.x + v.y + v.z + v.w;
    ss += v.x * v.x + v.y * v.y + v.z * v.z + v.w * v.w;
  }
  for (int m = 16; m >= 1; m >>= 1) {
    s += __shfl_xor(s, m);
    ss += __shfl_xor(ss, m);
  }
  __shared__ float red[16];
  int wave = tid >> 5, lane = tid & 31;
  if (lane == 0) { red[wave] = s; red[8 + wave] = ss; }
  __syncthreads();
  float S = 0.f, SS = 0.f;
  for (int w = 0; w < 8; ++w) { S += red[w]; SS += red[8 + w]; }
  float inv_n = 1.f / (float)n;
  float mean = S * inv_n;
  float var = SS * inv_n - mean * mean;
  float rstd = rsqrtf(var + 1e-5f);
  for (int cl = 0; cl < cpg; ++cl) {
    int c = g * cpg + cl;
    float gm = gamma[c] * rstd;
    float bt = beta[c] - mean * gm;
    const float* row = p + (size_t)cl * T;
    size_t obase = base + (size_t)cl * T;
    for (int t = tid * 4; t < T; t += 1024) {
      float4 v = *(const float4*)(row + t);
      v.x = v.x * gm + bt;
      v.y = v.y * gm + bt;
      v.z = v.z * gm + bt;
      v.w = v.w * gm + bt;
      if (doSilu) {
        v.x = v.x / (1.f + __expf(-v.x));
        v.y = v.y / (1.f + __expf(-v.y));
        v.z = v.z / (1.f + __expf(-v.z));
        v.w = v.w / (1.f + __expf(-v.w));
      }
      if (outF) *(float4*)(outF + obase + t) = v;
      if (outB) {
        bf16x4 bv;
        bv[0] = (__bf16)v.x; bv[1] = (__bf16)v.y;
        bv[2] = (__bf16)v.z; bv[3] = (__bf16)v.w;
        *(bf16x4*)(outB + obase + t) = bv;
      }
    }
  }
}

// ---------------- conv1d (k=1 or 3) as WMMA GEMM
// act (B,Cin,T) bf16 ; wgt (taps,Cout,Cin) bf16 ; out = W*act + bias (+resid)
// Block: 256 threads (8 waves). Tile: M=128 (Cout), N=64 (T). K step 32.
// Weights staged via Tensor Data Mover (one 128x32 tile per tap);
// activations staged by the waves with a +/-1 halo (Bs row r <=> t=n0-1+r).
__global__ __launch_bounds__(256) void conv_gemm_kernel(
    const __bf16* __restrict__ act, const __bf16* __restrict__ wgt,
    const float* __restrict__ bias, const float* __restrict__ resid,
    float* __restrict__ outF, __bf16* __restrict__ outB, int Cin, int Cout,
    int T, int taps) {
  __shared__ __bf16 As[3 * 128 * 32];  // [tap][m][k]
  __shared__ __bf16 Bs[66 * 32];       // [n+1][k], halo rows 0 and 65
  int b = blockIdx.z;
  int m0 = blockIdx.y * 128;
  int n0 = blockIdx.x * 64;
  int tid = threadIdx.x;
  int lane = tid & 31, wave = tid >> 5;
  int nn = lane & 15;
  int khalf16 = (lane >> 4) * 16;
  int base8 = (lane >> 4) * 8;
  int tapc = (taps == 3) ? 1 : 0;
  unsigned asBase = (unsigned)(uintptr_t)(void*)&As[0];

  f32x8 acc[4];
  for (int j = 0; j < 4; ++j)
    for (int r = 0; r < 8; ++r) acc[j][r] = 0.f;

  const __bf16* actB = act + (size_t)b * Cin * T;

  int bk = tid & 31;        // 0..31
  int bn = (tid >> 5) * 8;  // 0..56

  for (int kc = 0; kc < Cin; kc += 32) {
    __syncthreads();
    // A tiles via TDM: one 128x32 bf16 tile per tap (waves 0..taps-1)
    if (wave < taps) {
      int tpw = wave < taps ? wave : 0;  // stays in-bounds even if EXEC==0 issue
      tdm_load_2d_bf16(asBase + (unsigned)tpw * 8192u,
                       wgt + ((size_t)tpw * Cout + m0) * Cin + kc, Cin, Cout, 32,
                       128, Cin);
    }
    // B interior rows 1..64 (t = n0..n0+63), unconditional vector path
    {
      const __bf16* src = actB + (size_t)(kc + bk) * T + n0 + bn;
      bf16x8 v = *(const bf16x8*)src;
      for (int i = 0; i < 8; ++i) Bs[(1 + bn + i) * 32 + bk] = v[i];
    }
    // B halo rows (block-uniform guards, wave 0 only)
    if (taps == 3 && tid < 32) {
      int k = tid;
      const __bf16* col = actB + (size_t)(kc + k) * T;
      Bs[k] = (n0 - 1 >= 0) ? col[n0 - 1] : (__bf16)0.f;
      Bs[65 * 32 + k] = (n0 + 64 < T) ? col[n0 + 64] : (__bf16)0.f;
    }
    __builtin_amdgcn_s_wait_tensorcnt(0);  // NOP for non-issuing waves
    __syncthreads();
    for (int tp = 0; tp < taps; ++tp) {
      bf16x16 af;
      {
        const __bf16* ap = &As[tp * 4096 + (wave * 16 + nn) * 32];
        bf16x8 lo = *(const bf16x8*)(ap + khalf16);
        bf16x8 hi = *(const bf16x8*)(ap + khalf16 + 8);
        for (int i = 0; i < 8; ++i) { af[i] = lo[i]; af[8 + i] = hi[i]; }
      }
      int rshift = 1 + tp - tapc;  // taps==3: 0,1,2 ; taps==1: 1
      for (int j = 0; j < 4; ++j) {
        bf16x16 bf_ =
            *(const bf16x16*)&Bs[(16 * j + nn + rshift) * 32 + khalf16];
        acc[j] = wmma_bf16(af, bf_, acc[j]);
      }
    }
  }

  int rbase = (lane >> 4) * 8;
  for (int j = 0; j < 4; ++j) {
    int gn = n0 + 16 * j + nn;
    for (int r = 0; r < 8; ++r) {
      int gm = m0 + wave * 16 + rbase + r;
      float v = acc[j][r] + bias[gm];
      size_t o = ((size_t)b * Cout + gm) * T + gn;
      if (resid) v += resid[o];
      if (outB) outB[o] = (__bf16)v;
      else outF[o] = v;
    }
  }
}

// ---------------- flash attention with WMMA (bf16 QK/PV, f32 softmax)
// qkv (B,3C,T) bf16, channel = h*192 + {0,64,128} + c ; out (B,C,T) bf16
__global__ __launch_bounds__(256) void flash_attn_kernel(
    const __bf16* __restrict__ qkv, const float* __restrict__ tab,
    __bf16* __restrict__ outA) {
  __shared__ __bf16 Qs[128 * 64];     // [t][c]
  __shared__ __bf16 Ks[64 * 64];      // [s][c]
  __shared__ __bf16 Vs[64 * 64];      // [c][s]
  __shared__ __bf16 Ps[8 * 16 * 64];  // per-wave P scratch [t][s]
  __shared__ float biasRow[2048];

  int q0 = blockIdx.x * 128;
  int h = blockIdx.y;
  int b = blockIdx.z;
  int tid = threadIdx.x;
  int lane = tid & 31, wave = tid >> 5;
  int nn = lane & 15;
  int khalf16 = (lane >> 4) * 16;
  int base8 = (lane >> 4) * 8;
  int rbase = base8;

  const __bf16* qptr = qkv + ((size_t)b * (3 * C_DIM) + (size_t)h * 192) * T_DIM;
  const __bf16* kptr = qptr + (size_t)64 * T_DIM;
  const __bf16* vptr = qptr + (size_t)128 * T_DIM;

  for (int i = tid * 4; i < 2048; i += 1024)
    *(float4*)&biasRow[i] = *(const float4*)&tab[h * 2048 + i];
  {  // stage Q tile (transpose (c,t)->[t][c])
    int c = tid & 63;
    int seg = tid >> 6;
    for (int it = 0; it < 2; ++it) {
      int mb = (seg + it * 4) * 16;
      bf16x16 v = *(const bf16x16*)(qptr + (size_t)c * T_DIM + q0 + mb);
      for (int i = 0; i < 16; ++i) Qs[(mb + i) * 64 + c] = v[i];
    }
  }
  __syncthreads();

  bf16x16 aq[2];
  {
    const __bf16* qp = &Qs[(wave * 16 + nn) * 64];
    for (int kk = 0; kk < 2; ++kk) {
      bf16x8 lo = *(const bf16x8*)(qp + kk * 32 + khalf16);
      bf16x8 hi = *(const bf16x8*)(qp + kk * 32 + khalf16 + 8);
      for (int i = 0; i < 8; ++i) { aq[kk][i] = lo[i]; aq[kk][8 + i] = hi[i]; }
    }
  }

  f32x8 oacc[4];
  float mrow[8], lrow[8];
  for (int j = 0; j < 4; ++j)
    for (int r = 0; r < 8; ++r) oacc[j][r] = 0.f;
  for (int r = 0; r < 8; ++r) { mrow[r] = -1e30f; lrow[r] = 0.f; }

  __bf16* Pw = &Ps[wave * 16 * 64];

  for (int kt0 = 0; kt0 < T_DIM; kt0 += 64) {
    __syncthreads();
    {  // stage K (transpose) and V (direct)
      int c = tid & 63;
      int sb = (tid >> 6) * 16;
      bf16x16 kv = *(const bf16x16*)(kptr + (size_t)c * T_DIM + kt0 + sb);
      for (int i = 0; i < 16; ++i) Ks[(sb + i) * 64 + c] = kv[i];
      int c2 = tid >> 2;
      int sb2 = (tid & 3) * 16;
      *(bf16x16*)&Vs[c2 * 64 + sb2] =
          *(const bf16x16*)(vptr + (size_t)c2 * T_DIM + kt0 + sb2);
    }
    __syncthreads();

    f32x8 sacc[4];
    for (int j = 0; j < 4; ++j)
      for (int r = 0; r < 8; ++r) sacc[j][r] = 0.f;
    for (int j = 0; j < 4; ++j)
      for (int kk = 0; kk < 2; ++kk) {
        bf16x16 bf_ = *(const bf16x16*)&Ks[(16 * j + nn) * 64 + kk * 32 + khalf16];
        sacc[j] = wmma_bf16(aq[kk], bf_, sacc[j]);
      }

    float tmax[8];
    for (int r = 0; r < 8; ++r) tmax[r] = -1e30f;
    int tq0 = q0 + wave * 16 + rbase;
    for (int j = 0; j < 4; ++j) {
      int sk = kt0 + 16 * j + nn;
      for (int r = 0; r < 8; ++r) {
        float val = sacc[j][r] * 0.125f + biasRow[tq0 + r - sk + 1024];
        sacc[j][r] = val;
        tmax[r] = fmaxf(tmax[r], val);
      }
    }
    for (int r = 0; r < 8; ++r)
      for (int m = 1; m < 16; m <<= 1) tmax[r] = fmaxf(tmax[r], __shfl_xor(tmax[r], m));

    float crr[8];
    for (int r = 0; r < 8; ++r) {
      float mn = fmaxf(mrow[r], tmax[r]);
      crr[r] = __expf(mrow[r] - mn);
      mrow[r] = mn;
      lrow[r] *= crr[r];
    }
    for (int j = 0; j < 4; ++j)
      for (int r = 0; r < 8; ++r) oacc[j][r] *= crr[r];

    for (int j = 0; j < 4; ++j)
      for (int r = 0; r < 8; ++r) {
        float pe = __expf(sacc[j][r] - mrow[r]);
        lrow[r] += pe;
        Pw[(rbase + r) * 64 + 16 * j + nn] = (__bf16)pe;
      }

    bf16x16 ap2[2];
    {
      const __bf16* pp = &Ps[(wave * 16 + nn) * 64];
      for (int kk = 0; kk < 2; ++kk) {
        bf16x8 lo = *(const bf16x8*)(pp + kk * 32 + khalf16);
        bf16x8 hi = *(const bf16x8*)(pp + kk * 32 + khalf16 + 8);
        for (int i = 0; i < 8; ++i) { ap2[kk][i] = lo[i]; ap2[kk][8 + i] = hi[i]; }
      }
    }
    for (int j2 = 0; j2 < 4; ++j2)
      for (int kk = 0; kk < 2; ++kk) {
        bf16x16 bv = *(const bf16x16*)&Vs[(16 * j2 + nn) * 64 + kk * 32 + khalf16];
        oacc[j2] = wmma_bf16(ap2[kk], bv, oacc[j2]);
      }
  }

  float rl[8];
  for (int r = 0; r < 8; ++r) {
    for (int m = 1; m < 16; m <<= 1) lrow[r] += __shfl_xor(lrow[r], m);
    rl[r] = 1.f / lrow[r];
  }

  for (int j2 = 0; j2 < 4; ++j2) {
    int c = 16 * j2 + nn;
    bf16x8 ov;
    for (int r = 0; r < 8; ++r) ov[r] = (__bf16)(oacc[j2][r] * rl[r]);
    *(bf16x8*)(outA + ((size_t)b * C_DIM + (size_t)(h * 64 + c)) * T_DIM + q0 +
               wave * 16 + rbase) = ov;
  }
}

extern "C" void kernel_launch(void* const* d_in, const int* in_sizes, int n_in,
                              void* d_out, int out_size, void* d_ws, size_t ws_size,
                              hipStream_t stream) {
  (void)in_sizes; (void)n_in; (void)out_size; (void)ws_size;
  const int B = 8, C = 1024, T = 1024, H = 16;

  const float* x = (const float*)d_in[0];

  char* ws = (char*)d_ws;
  size_t off = 0;
  auto alloc = [&](size_t bytes) -> char* {
    char* p = ws + off;
    off += (bytes + 255) & ~(size_t)255;
    return p;
  };
  __bf16* wb1  = (__bf16*)alloc((size_t)3 * C * C * 2);
  __bf16* wb2  = (__bf16*)alloc((size_t)3 * C * C * 2);
  __bf16* wq   = (__bf16*)alloc((size_t)3 * C * C * 2);
  __bf16* wp   = (__bf16*)alloc((size_t)C * C * 2);
  float*  tab  = (float*)alloc((size_t)H * 2048 * 4);
  __bf16* actb = (__bf16*)alloc((size_t)B * C * T * 2);
  float*  ftmp = (float*)alloc((size_t)B * C * T * 4);
  float*  xin  = (float*)alloc((size_t)B * C * T * 4);
  __bf16* qkvb = (__bf16*)alloc((size_t)B * 3 * C * T * 2);

  // weight conversion + bias table
  wcvt_kernel<<<(3 * C * C + 255) / 256, 256, 0, stream>>>((const float*)d_in[3], wb1, C, C, 3);
  wcvt_kernel<<<(3 * C * C + 255) / 256, 256, 0, stream>>>((const float*)d_in[7], wb2, C, C, 3);
  wcvt_kernel<<<(3 * C * C + 255) / 256, 256, 0, stream>>>((const float*)d_in[13], wq, 3 * C, C, 1);
  wcvt_kernel<<<(C * C + 255) / 256, 256, 0, stream>>>((const float*)d_in[15], wp, C, C, 1);
  btab_kernel<<<(H * 2048 + 255) / 256, 256, 0, stream>>>((const float*)d_in[17], tab);

  // h = silu(gn32(x))
  gn_kernel<<<dim3(32, B), 256, 0, stream>>>(x, (const float*)d_in[1], (const float*)d_in[2],
                                             nullptr, actb, C, T, 32, 1);
  // f1 = conv1(h)
  conv_gemm_kernel<<<dim3(T / 64, C / 128, B), 256, 0, stream>>>(
      actb, wb1, (const float*)d_in[4], nullptr, ftmp, nullptr, C, C, T, 3);
  // h = silu(gn32(f1))
  gn_kernel<<<dim3(32, B), 256, 0, stream>>>(ftmp, (const float*)d_in[5], (const float*)d_in[6],
                                             nullptr, actb, C, T, 32, 1);
  // f2 = conv2(h) + x
  conv_gemm_kernel<<<dim3(T / 64, C / 128, B), 256, 0, stream>>>(
      actb, wb2, (const float*)d_in[8], x, ftmp, nullptr, C, C, T, 3);
  // xin = gn8(f2)
  gn_kernel<<<dim3(8, B), 256, 0, stream>>>(ftmp, (const float*)d_in[9], (const float*)d_in[10],
                                            xin, nullptr, C, T, 8, 0);
  // hb = gn32(xin)
  gn_kernel<<<dim3(32, B), 256, 0, stream>>>(xin, (const float*)d_in[11], (const float*)d_in[12],
                                             nullptr, actb, C, T, 32, 0);
  // qkv = qkv_w * hb + qkv_b  (bf16 out)
  conv_gemm_kernel<<<dim3(T / 64, (3 * C) / 128, B), 256, 0, stream>>>(
      actb, wq, (const float*)d_in[14], nullptr, nullptr, qkvb, C, 3 * C, T, 1);
  // a = attention(q,k,v) with rel-pos bias (bf16 out)
  flash_attn_kernel<<<dim3(T / 128, H, B), 256, 0, stream>>>(qkvb, tab, actb);
  // out = xin + proj(a)
  conv_gemm_kernel<<<dim3(T / 64, C / 128, B), 256, 0, stream>>>(
      actb, wp, (const float*)d_in[16], xin, (float*)d_out, nullptr, C, C, T, 1);
}
